// CatMarginalHead_30983894073666
// MI455X (gfx1250) — matrix operand order repaired and checked by reading, not compiled, not run on hardware
//
#include <hip/hip_runtime.h>
#include <hip/hip_bf16.h>
#include <math.h>

// ---- problem constants (from reference) ----
#define B_ROWS   8192
#define N_SLOTS  12
#define H_DIM    512
#define TWO_H    1024
#define V_DIM    256
#define LN_EPS   1e-5f

// ---- tiling ----
#define BM       32          // rows (batch) per block
#define KT_STEP  32          // K per WMMA
#define NUM_KT   (TWO_H / KT_STEP)   // 32
#define NUM_NT   (V_DIM / 16)        // 16 column tiles
#define ACT_PAD  8                   // f16 pad -> row stride 1032 (2064B, +4 banks/row)
#define ACT_STRIDE (TWO_H + ACT_PAD) // 1032

typedef _Float16 v8h  __attribute__((ext_vector_type(8)));
typedef _Float16 v16h __attribute__((ext_vector_type(16)));
typedef float    v8f  __attribute__((ext_vector_type(8)));

// =====================================================================
// Kernel 1: convert pred_W (f32, [N][2H][V]) into f16 WMMA B-fragments.
// Fragment layout per (n, kt, nt): 32 lanes x 16 f16, where lane's i-th
// element = W[K = kt*32 + 16*(lane>=16) + i][Vcol = nt*16 + (lane&15)].
// Contiguous k order (k = 16h + i); fused_kernel builds its A fragments in the same k order.
// Staged through LDS so global read AND write are coalesced.
// grid = N_SLOTS*NUM_KT blocks of 256 threads.
// =====================================================================
__global__ __launch_bounds__(256) void convW_kernel(const float* __restrict__ W,
                                                    _Float16* __restrict__ wsW) {
    __shared__ float tile[KT_STEP][V_DIM];   // 32 KB
    const int blk = blockIdx.x;              // = n*NUM_KT + kt
    const int n  = blk >> 5;
    const int kt = blk & 31;

    const float* src = W + ((size_t)n * TWO_H + (size_t)kt * KT_STEP) * V_DIM;
    for (int idx = threadIdx.x; idx < KT_STEP * V_DIM; idx += 256)
        tile[idx >> 8][idx & 255] = src[idx];
    __syncthreads();

    _Float16* dst = wsW + (size_t)blk * (NUM_NT * 32 * 16);   // 8192 f16 per (n,kt)
    for (int e = threadIdx.x; e < NUM_NT * 32 * 16; e += 256) {
        const int i    = e & 15;
        const int lane = (e >> 4) & 31;
        const int nt   = e >> 9;
        const int krow = ((lane >> 4) << 4) + i;      // 0..31 within kt chunk
        const int vcol = nt * 16 + (lane & 15);
        dst[e] = (_Float16)tile[krow][vcol];
    }
}

// =====================================================================
// Kernel 2: fused gather + exclusive-cumsum + concat + LayerNorm + GeLU
//           + per-slot GEMM (WMMA f16 -> f32) + bias.
// Block: 512 threads (16 wave32), BM=32 rows. Wave w owns rows 2w,2w+1
// for the activation stage and column tile nt=w (+ both 16-row M-tiles)
// for the GEMM stage.
// =====================================================================
__device__ __forceinline__ float gelu_exact(float x) {
    return 0.5f * x * (1.0f + erff(x * 0.70710678118654752f));
}

__global__ __launch_bounds__(512) void fused_kernel(
    const float* __restrict__ ctxIn,      // [B,H]
    const int*   __restrict__ feats,      // [B,N]
    const float* __restrict__ embT,       // [N,V,H]
    const float* __restrict__ gamma,      // [2H]
    const float* __restrict__ beta,       // [2H]
    const float* __restrict__ predb,      // [N,V]
    const _Float16* __restrict__ wsW,     // swizzled f16 fragments
    float* __restrict__ out)              // [B,N,V] f32
{
    extern __shared__ __align__(16) char smem[];
    _Float16 (*sAct)[ACT_STRIDE] = (_Float16 (*)[ACT_STRIDE])smem;          // 66048 B
    float* sG = (float*)(smem + (size_t)BM * ACT_STRIDE * sizeof(_Float16)); // 4096 B
    float* sB = sG + TWO_H;                                                  // 4096 B

    const int tid  = threadIdx.x;
    const int lane = tid & 31;
    const int w    = tid >> 5;            // 0..15
    const int b0   = blockIdx.x * BM;

    for (int i = tid; i < TWO_H; i += 512) { sG[i] = gamma[i]; sB[i] = beta[i]; }

    // activation-stage rows for this wave; per-lane h-chunk [hb, hb+16)
    const int r0 = 2 * w, r1 = 2 * w + 1;
    const int hb = lane * 16;

    float ctx0[16], ctx1[16], pre0[16], pre1[16];
    {
        const float* c0 = ctxIn + (size_t)(b0 + r0) * H_DIM + hb;
        const float* c1 = ctxIn + (size_t)(b0 + r1) * H_DIM + hb;
        #pragma unroll
        for (int i = 0; i < 16; i++) {
            ctx0[i] = c0[i]; ctx1[i] = c1[i];
            pre0[i] = 0.0f;  pre1[i] = 0.0f;
        }
    }
    __syncthreads();

    for (int n = 0; n < N_SLOTS; n++) {
        // ---------- LayerNorm + GeLU -> f16 LDS tile ----------
        {
            float s0 = 0.f, q0 = 0.f, s1 = 0.f, q1 = 0.f;
            #pragma unroll
            for (int i = 0; i < 16; i++) {
                s0 += ctx0[i] + pre0[i];
                q0 += ctx0[i] * ctx0[i] + pre0[i] * pre0[i];
                s1 += ctx1[i] + pre1[i];
                q1 += ctx1[i] * ctx1[i] + pre1[i] * pre1[i];
            }
            #pragma unroll
            for (int off = 16; off >= 1; off >>= 1) {
                s0 += __shfl_xor(s0, off, 32); q0 += __shfl_xor(q0, off, 32);
                s1 += __shfl_xor(s1, off, 32); q1 += __shfl_xor(q1, off, 32);
            }
            const float mu0 = s0 * (1.f / TWO_H);
            const float mu1 = s1 * (1.f / TWO_H);
            const float rs0 = rsqrtf(q0 * (1.f / TWO_H) - mu0 * mu0 + LN_EPS);
            const float rs1 = rsqrtf(q1 * (1.f / TWO_H) - mu1 * mu1 + LN_EPS);

            v8h oc0a, oc0b, op0a, op0b, oc1a, oc1b, op1a, op1b;
            #pragma unroll
            for (int i = 0; i < 16; i++) {
                const float gc = sG[hb + i],        bc = sB[hb + i];
                const float gp = sG[H_DIM + hb + i], bp = sB[H_DIM + hb + i];
                const float a0 = gelu_exact((ctx0[i] - mu0) * rs0 * gc + bc);
                const float p0 = gelu_exact((pre0[i] - mu0) * rs0 * gp + bp);
                const float a1 = gelu_exact((ctx1[i] - mu1) * rs1 * gc + bc);
                const float p1 = gelu_exact((pre1[i] - mu1) * rs1 * gp + bp);
                if (i < 8) { oc0a[i] = (_Float16)a0; op0a[i] = (_Float16)p0;
                             oc1a[i] = (_Float16)a1; op1a[i] = (_Float16)p1; }
                else       { oc0b[i-8] = (_Float16)a0; op0b[i-8] = (_Float16)p0;
                             oc1b[i-8] = (_Float16)a1; op1b[i-8] = (_Float16)p1; }
            }
            *(v8h*)&sAct[r0][hb]             = oc0a; *(v8h*)&sAct[r0][hb + 8]         = oc0b;
            *(v8h*)&sAct[r0][H_DIM + hb]     = op0a; *(v8h*)&sAct[r0][H_DIM + hb + 8] = op0b;
            *(v8h*)&sAct[r1][hb]             = oc1a; *(v8h*)&sAct[r1][hb + 8]         = oc1b;
            *(v8h*)&sAct[r1][H_DIM + hb]     = op1a; *(v8h*)&sAct[r1][H_DIM + hb + 8] = op1b;
        }
        __syncthreads();

        // ---------- GEMM: [32 x 1024] x [1024 x 16] for column tile nt=w ----------
        {
            v8f acc0 = {}; v8f acc1 = {};
            const _Float16* wn = wsW + ((size_t)(n * NUM_KT) * NUM_NT + w) * (32 * 16)
                                     + lane * 16;
            const int koff = (lane >> 4) ? 16 : 0;    // A-fragment per-lane K base
            const int Mlo = (lane & 15), Mhi = (lane & 15) + 16;

            #pragma unroll 8
            for (int kt = 0; kt < NUM_KT; kt++) {
                const v16h bv = *(const v16h*)(wn + (size_t)kt * (NUM_NT * 32 * 16));
                const int kb = kt * KT_STEP + koff;
                const v8h l0 = *(const v8h*)&sAct[Mlo][kb];
                const v8h h0 = *(const v8h*)&sAct[Mlo][kb + 8];
                const v8h l1 = *(const v8h*)&sAct[Mhi][kb];
                const v8h h1 = *(const v8h*)&sAct[Mhi][kb + 8];
                const v16h a0 = __builtin_shufflevector(l0, h0, 0,1,2,3,4,5,6,7,8,9,10,11,12,13,14,15);
                const v16h a1 = __builtin_shufflevector(l1, h1, 0,1,2,3,4,5,6,7,8,9,10,11,12,13,14,15);
                acc0 = __builtin_amdgcn_wmma_f32_16x16x32_f16(false, a0, false, bv,
                                                              (short)0, acc0, false, false);
                acc1 = __builtin_amdgcn_wmma_f32_16x16x32_f16(false, a1, false, bv,
                                                              (short)0, acc1, false, false);
            }

            const float bias = predb[n * V_DIM + w * 16 + (lane & 15)];
            // C layout: VGPR i -> M = i + 8*(lane>=16); N = lane&15 within tile
            const size_t obase = ((size_t)b0 * N_SLOTS + n) * V_DIM + w * 16 + (lane & 15);
            const int mrow = (lane >> 4) * 8;
            #pragma unroll
            for (int i = 0; i < 8; i++) {
                out[obase + (size_t)(mrow + i)      * (N_SLOTS * V_DIM)] = acc0[i] + bias;
                out[obase + (size_t)(mrow + i + 16) * (N_SLOTS * V_DIM)] = acc1[i] + bias;
            }
        }

        // ---------- prefix += emb_tables[n, feats[b,n], :] (registers only) ----------
        {
            const int f0 = feats[(b0 + r0) * N_SLOTS + n];
            const int f1 = feats[(b0 + r1) * N_SLOTS + n];
            const float* e0 = embT + ((size_t)n * V_DIM + f0) * H_DIM + hb;
            const float* e1 = embT + ((size_t)n * V_DIM + f1) * H_DIM + hb;
            #pragma unroll
            for (int i = 0; i < 16; i++) { pre0[i] += e0[i]; pre1[i] += e1[i]; }
        }
        __syncthreads();   // sAct fully consumed before next n overwrites it
    }
}

// =====================================================================
extern "C" void kernel_launch(void* const* d_in, const int* in_sizes, int n_in,
                              void* d_out, int out_size, void* d_ws, size_t ws_size,
                              hipStream_t stream) {
    const float* ctx   = (const float*)d_in[0];   // input_embedding [B,H]
    const int*   feats = (const int*)  d_in[1];   // features        [B,N]
    const float* embT  = (const float*)d_in[2];   // emb_tables      [N,V,H]
    const float* gamma = (const float*)d_in[3];   // ln_gamma        [2H]
    const float* beta  = (const float*)d_in[4];   // ln_beta         [2H]
    const float* predW = (const float*)d_in[5];   // pred_W          [N,2H,V]
    const float* predb = (const float*)d_in[6];   // pred_b          [N,V]
    float* out = (float*)d_out;
    _Float16* wsW = (_Float16*)d_ws;              // 12*32*16*512 f16 = 6.29 MB

    // 1) one-shot weight swizzle f32 -> f16 B-fragments (L2-resident afterwards)
    convW_kernel<<<N_SLOTS * NUM_KT, 256, 0, stream>>>(predW, wsW);

    // 2) fused pipeline
    const size_t shmem = (size_t)BM * ACT_STRIDE * sizeof(_Float16)   // 66048
                       + 2 * TWO_H * sizeof(float);                   // + 8192 = 74240
    hipFuncSetAttribute((const void*)fused_kernel,
                        hipFuncAttributeMaxDynamicSharedMemorySize, (int)shmem);
    fused_kernel<<<B_ROWS / BM, 512, shmem, stream>>>(
        ctx, feats, embT, gamma, beta, predb, wsW, out);
}
